// InteractionBlock_27994596835751
// MI455X (gfx1250) — matrix operand order repaired and checked by reading, not compiled, not run on hardware
//
#include <hip/hip_runtime.h>
#include <hip/hip_bf16.h>
#include <hip/hip_fp16.h>

typedef __attribute__((ext_vector_type(16))) _Float16 v16h;
typedef __attribute__((ext_vector_type(8)))  float    v8f;

#define Hdim 128
#define BDdim 8
#define NRdim 6
#define NSdim 7

__device__ __forceinline__ float silu_f(float x) { return x / (1.0f + __expf(-x)); }

// ---------------- weight packing into CDNA5 WMMA B-fragment order ----------------
// B layout (16-bit, 32x16 tile, wave32): element j of v16h held by lane L is
//   K = (j<8 ? (L>>4)*8 + j : 16 + (L>>4)*8 + j-8),  N = (L&15).   Packed index: ((t*nsteps + s)*32 + lane)*16 + j
__global__ void k_pack128(const float* __restrict__ W, _Float16* __restrict__ out) {
  int idx = blockIdx.x * blockDim.x + threadIdx.x;
  if (idx >= 8 * 4 * 32 * 16) return;
  int j = idx & 15, lane = (idx >> 4) & 31, s = (idx >> 9) & 3, t = (idx >> 11) & 7;
  int k = s * 32 + (lane >> 4) * 8 + (j & 7) + (j >> 3) * 16;
  int n = t * 16 + (lane & 15);
  out[idx] = (_Float16)W[k * Hdim + n];
}

// Wb is [i=128][b=8][h=128]; we need Wb_r[k = b*128+h][n = i]
__global__ void k_packWb(const float* __restrict__ Wb, _Float16* __restrict__ out) {
  int idx = blockIdx.x * blockDim.x + threadIdx.x;
  if (idx >= 8 * 32 * 32 * 16) return;
  int j = idx & 15, lane = (idx >> 4) & 31, s = (idx >> 9) & 31, t = (idx >> 14) & 7;
  int k = s * 32 + (lane >> 4) * 8 + (j & 7) + (j >> 3) * 16;   // 0..1023
  int n = t * 16 + (lane & 15);            // output channel i
  out[idx] = (_Float16)Wb[(size_t)n * 1024 + k];
}

// ---------------- small dense prologue (memory-bound, VALU) ----------------
__global__ void k_dist(const float* __restrict__ xd, const float* __restrict__ Wd,
                       float* __restrict__ D, int E) {
  int tid = blockIdx.x * blockDim.x + threadIdx.x;
  if (tid >= E * Hdim) return;
  int e = tid >> 7, h = tid & 127;
  const float* xr = xd + (size_t)e * NRdim;
  float v = 0.0f;
#pragma unroll
  for (int r = 0; r < NRdim; ++r) v += xr[r] * Wd[r * Hdim + h];
  D[tid] = v;
}

__global__ void k_avals(const float* __restrict__ xa, const float* __restrict__ Wa,
                        float* __restrict__ Av, int E) {
  int tid = blockIdx.x * blockDim.x + threadIdx.x;
  if (tid >= E * BDdim) return;
  int e = tid >> 3, b = tid & 7;
  const float* xr = xa + (size_t)e * (NSdim * NRdim);
  float v = 0.0f;
#pragma unroll
  for (int j = 0; j < NSdim * NRdim; ++j) v += xr[j] * Wa[j * BDdim + b];
  Av[tid] = v;
}

__global__ void k_zero(float* __restrict__ p, int n) {
  int tid = blockIdx.x * blockDim.x + threadIdx.x;
  if (tid < n) p[tid] = 0.0f;
}

// S[tgt] += P[src]  (L2-resident float atomics)
__global__ void k_scatter(const float* __restrict__ P, const int* __restrict__ idx,
                          float* __restrict__ S, int A) {
  int tid = blockIdx.x * blockDim.x + threadIdx.x;
  if (tid >= A * Hdim) return;
  int a = tid >> 7, h = tid & 127;
  int s = idx[a];        // src row
  int t = idx[A + a];    // tgt row
  atomicAdd(&S[(size_t)t * Hdim + h], P[(size_t)s * Hdim + h]);
}

__global__ void k_mul_inplace(float* __restrict__ a, const float* __restrict__ b, int n) {
  int tid = blockIdx.x * blockDim.x + threadIdx.x;
  if (tid < n) a[tid] *= b[tid];
}

// ---------------- fused WMMA GEMM: Y = [addsrc +] [silu](X @ W + bias) ----------------
// block = 256 threads = 8 wave32s; each wave computes a 16x128 output strip.
__global__ __launch_bounds__(256) void k_gemm128(
    const float* __restrict__ X, const _Float16* __restrict__ Wp,
    const float* __restrict__ bias, const float* __restrict__ addsrc,
    float* __restrict__ Y, int E, int do_silu) {
  __shared__ alignas(32) _Float16 sW[8 * 4 * 32 * 16];  // 32 KB packed weights
  {
    const uint4* s4 = (const uint4*)Wp;
    uint4* d4 = (uint4*)sW;
    for (int i = threadIdx.x; i < (8 * 4 * 32 * 16) / 8; i += 256) d4[i] = s4[i];
  }
  __syncthreads();

  const int lane = threadIdx.x & 31;
  const int wave = threadIdx.x >> 5;
  const int l15 = lane & 15;
  const int hi = lane >> 4;
  const int row_base = blockIdx.x * 128 + wave * 16;
  int mrow = row_base + l15;
  if (mrow >= E) mrow = E - 1;
  const float* xr = X + (size_t)mrow * Hdim;

  v8f acc[8] = {};
#pragma unroll
  for (int s = 0; s < 4; ++s) {
    const int kb = s * 32 + hi * 8;
    const float4 c0 = *(const float4*)(xr + kb);
    const float4 c1 = *(const float4*)(xr + kb + 4);
    const float4 c2 = *(const float4*)(xr + kb + 16);
    const float4 c3 = *(const float4*)(xr + kb + 20);
    v16h a;
    a[0] = (_Float16)c0.x;  a[1] = (_Float16)c0.y;  a[2] = (_Float16)c0.z;  a[3] = (_Float16)c0.w;
    a[4] = (_Float16)c1.x;  a[5] = (_Float16)c1.y;  a[6] = (_Float16)c1.z;  a[7] = (_Float16)c1.w;
    a[8] = (_Float16)c2.x;  a[9] = (_Float16)c2.y;  a[10] = (_Float16)c2.z; a[11] = (_Float16)c2.w;
    a[12] = (_Float16)c3.x; a[13] = (_Float16)c3.y; a[14] = (_Float16)c3.z; a[15] = (_Float16)c3.w;
#pragma unroll
    for (int t = 0; t < 8; ++t) {
      const v16h b = *(const v16h*)(sW + (size_t)((t * 4 + s) * 32 + lane) * 16);
      acc[t] = __builtin_amdgcn_wmma_f32_16x16x32_f16(false, a, false, b, (short)0,
                                                      acc[t], false, false);
    }
  }

#pragma unroll
  for (int t = 0; t < 8; ++t) {
    const int n = t * 16 + l15;
    const float bv = bias ? bias[n] : 0.0f;
#pragma unroll
    for (int r = 0; r < 8; ++r) {
      const int m = row_base + r + hi * 8;   // C/D layout: VGPR r -> M=r (+8 for hi lanes)
      if (m < E) {
        float v = acc[t][r] + bv;
        if (do_silu) v = silu_f(v);
        if (addsrc) v += addsrc[(size_t)m * Hdim + n];
        Y[(size_t)m * Hdim + n] = v;
      }
    }
  }
}

// ---------------- bilinear GEMM: agg = T @ Wb_r,  T[e, b*128+h] = Av[e,b]*U[e,h] ----------------
__global__ __launch_bounds__(256) void k_gemm_bilinear(
    const float* __restrict__ U, const float* __restrict__ Av,
    const _Float16* __restrict__ Wp, float* __restrict__ Y, int E) {
  const int lane = threadIdx.x & 31;
  const int wave = threadIdx.x >> 5;
  const int l15 = lane & 15;
  const int hi = lane >> 4;
  const int row_base = blockIdx.x * 128 + wave * 16;
  int mrow = row_base + l15;
  if (mrow >= E) mrow = E - 1;
  const float* ur = U + (size_t)mrow * Hdim;
  const float* ar = Av + (size_t)mrow * BDdim;

  v8f acc[8] = {};
#pragma unroll 4
  for (int s = 0; s < 32; ++s) {            // K = 1024, 32 per step; b = s>>2 fixed per step
    const float asc = ar[s >> 2];
    const int kb = (s & 3) * 32 + hi * 8;
    const float4 c0 = *(const float4*)(ur + kb);
    const float4 c1 = *(const float4*)(ur + kb + 4);
    const float4 c2 = *(const float4*)(ur + kb + 16);
    const float4 c3 = *(const float4*)(ur + kb + 20);
    v16h a;
    a[0] = (_Float16)(asc * c0.x);  a[1] = (_Float16)(asc * c0.y);
    a[2] = (_Float16)(asc * c0.z);  a[3] = (_Float16)(asc * c0.w);
    a[4] = (_Float16)(asc * c1.x);  a[5] = (_Float16)(asc * c1.y);
    a[6] = (_Float16)(asc * c1.z);  a[7] = (_Float16)(asc * c1.w);
    a[8] = (_Float16)(asc * c2.x);  a[9] = (_Float16)(asc * c2.y);
    a[10] = (_Float16)(asc * c2.z); a[11] = (_Float16)(asc * c2.w);
    a[12] = (_Float16)(asc * c3.x); a[13] = (_Float16)(asc * c3.y);
    a[14] = (_Float16)(asc * c3.z); a[15] = (_Float16)(asc * c3.w);
#pragma unroll
    for (int t = 0; t < 8; ++t) {
      const v16h b = *(const v16h*)(Wp + (size_t)((t * 32 + s) * 32 + lane) * 16);
      acc[t] = __builtin_amdgcn_wmma_f32_16x16x32_f16(false, a, false, b, (short)0,
                                                      acc[t], false, false);
    }
  }

#pragma unroll
  for (int t = 0; t < 8; ++t) {
    const int n = t * 16 + l15;
#pragma unroll
    for (int r = 0; r < 8; ++r) {
      const int m = row_base + r + hi * 8;
      if (m < E) Y[(size_t)m * Hdim + n] = acc[t][r];
    }
  }
}

extern "C" void kernel_launch(void* const* d_in, const int* in_sizes, int n_in,
                              void* d_out, int out_size, void* d_ws, size_t ws_size,
                              hipStream_t stream) {
  const float* x_dist  = (const float*)d_in[0];
  const float* x_angle = (const float*)d_in[1];
  const float* msg     = (const float*)d_in[2];
  const int*   aidx    = (const int*)d_in[3];
  const float* Wd = (const float*)d_in[4];
  const float* Wa = (const float*)d_in[5];
  const float* Ws = (const float*)d_in[6];
  const float* bs = (const float*)d_in[7];
  const float* Wt = (const float*)d_in[8];
  const float* bt = (const float*)d_in[9];
  const float* Wb = (const float*)d_in[10];
  const float* rb_w1 = (const float*)d_in[11];  const float* rb_b1 = (const float*)d_in[12];
  const float* rb_w2 = (const float*)d_in[13];  const float* rb_b2 = (const float*)d_in[14];
  const float* Wskip = (const float*)d_in[15];  const float* bskip = (const float*)d_in[16];
  const float* ra1_w1 = (const float*)d_in[17]; const float* ra1_b1 = (const float*)d_in[18];
  const float* ra1_w2 = (const float*)d_in[19]; const float* ra1_b2 = (const float*)d_in[20];
  const float* ra2_w1 = (const float*)d_in[21]; const float* ra2_b1 = (const float*)d_in[22];
  const float* ra2_w2 = (const float*)d_in[23]; const float* ra2_b2 = (const float*)d_in[24];

  const int E = in_sizes[0] / NRdim;
  const int A = in_sizes[3] / 2;
  if (E <= 0) return;

  // workspace carve (packed f16 weights + 3 rotating E x 128 f32 buffers + a-values)
  char* w = (char*)d_ws;
  auto take = [&](size_t bytes) -> char* {
    char* p = w; w += (bytes + 255) & ~(size_t)255; return p;
  };
  _Float16* pW[9];
  for (int i = 0; i < 9; ++i) pW[i] = (_Float16*)take((size_t)8 * 4 * 32 * 16 * sizeof(_Float16));
  _Float16* pWb = (_Float16*)take((size_t)8 * 32 * 32 * 16 * sizeof(_Float16));
  float* B0 = (float*)take((size_t)E * Hdim * sizeof(float));
  float* B1 = (float*)take((size_t)E * Hdim * sizeof(float));
  float* B2 = (float*)take((size_t)E * Hdim * sizeof(float));
  float* Av = (float*)take((size_t)E * BDdim * sizeof(float));
  (void)ws_size; (void)n_in; (void)out_size;

  const float* wsrc[9] = {Ws, Wt, rb_w1, rb_w2, Wskip, ra1_w1, ra1_w2, ra2_w1, ra2_w2};
  for (int i = 0; i < 9; ++i) k_pack128<<<64, 256, 0, stream>>>(wsrc[i], pW[i]);
  k_packWb<<<512, 256, 0, stream>>>(Wb, pWb);

  const int nEH = E * Hdim;
  const int gb = (E + 127) / 128;
  k_dist<<<(nEH + 255) / 256, 256, 0, stream>>>(x_dist, Wd, B0, E);            // d -> B0
  k_avals<<<(E * BDdim + 255) / 256, 256, 0, stream>>>(x_angle, Wa, Av, E);    // a -> Av

  k_gemm128<<<gb, 256, 0, stream>>>(msg, pW[0], bs, nullptr, B1, E, 0);        // P = msg@Ws+bs -> B1
  k_zero<<<(nEH + 255) / 256, 256, 0, stream>>>(B2, nEH);                      // S = 0
  k_scatter<<<(A * Hdim + 255) / 256, 256, 0, stream>>>(B1, aidx, B2, A);      // S[tgt]+=P[src]
  k_mul_inplace<<<(nEH + 255) / 256, 256, 0, stream>>>(B0, B2, nEH);           // u = d*S -> B0

  k_gemm_bilinear<<<gb, 256, 0, stream>>>(B0, Av, pWb, B1, E);                 // agg -> B1
  k_gemm128<<<gb, 256, 0, stream>>>(msg, pW[1], bt, B1, B2, E, 0);             // x1 = agg + msg@Wt+bt
  k_gemm128<<<gb, 256, 0, stream>>>(B2, pW[2], rb_b1, nullptr, B0, E, 1);      // h  = silu(x1@w1+b1)
  k_gemm128<<<gb, 256, 0, stream>>>(B0, pW[3], rb_b2, B2, B1, E, 1);           // x2 = x1 + silu(h@w2+b2)
  k_gemm128<<<gb, 256, 0, stream>>>(B1, pW[4], bskip, msg, B2, E, 1);          // x3 = msg + silu(x2@Wsk+b)
  k_gemm128<<<gb, 256, 0, stream>>>(B2, pW[5], ra1_b1, nullptr, B0, E, 1);     // h2
  k_gemm128<<<gb, 256, 0, stream>>>(B0, pW[6], ra1_b2, B2, B1, E, 1);          // x4 = x3 + silu(h2@..)
  k_gemm128<<<gb, 256, 0, stream>>>(B1, pW[7], ra2_b1, nullptr, B0, E, 1);     // h3
  k_gemm128<<<gb, 256, 0, stream>>>(B0, pW[8], ra2_b2, B1, (float*)d_out, E, 1); // out = x4 + silu(h3@..)
}
